// Attention_66468913873023
// MI455X (gfx1250) — matrix operand order repaired and checked by reading, not compiled, not run on hardware
//
#include <hip/hip_runtime.h>

// CDNA5 / gfx1250 cosine-similarity (Swin-v2) attention.
// B=256, nH=8, N=256, d=64. One workgroup per (b,h); 16 waves x 16 rows.
// HBM-bound (~540MB @ 23.3TB/s); all intermediates stay in LDS/VGPRs,
// matmuls use v_wmma_f32_16x16x32_f16 with f32 accumulation.

typedef _Float16 v16h __attribute__((ext_vector_type(16)));
typedef _Float16 v8h  __attribute__((ext_vector_type(8)));
typedef _Float16 v4h  __attribute__((ext_vector_type(4)));
typedef float    v8f  __attribute__((ext_vector_type(8)));
typedef float    v4f  __attribute__((ext_vector_type(4)));

#define NH 8
#define NN 256
#define DD 64
#define QK_STRIDE 80    // halves per row (160B, 32B aligned, conflict-skewed)
#define VT_STRIDE 272   // halves per row of v^T (544B, 32B aligned)
#define PT_STRIDE 48    // halves per row of per-wave P scratch (96B)
#define LOG100 4.605170185988092f
#define SMEM_BYTES ((2*NN*QK_STRIDE + DD*VT_STRIDE + 16*16*PT_STRIDE) * 2)

union Frag { v16h v; v8h h8[2]; };

extern "C" __global__ __launch_bounds__(512, 1)
void attn_cos_kernel(const float* __restrict__ q, const float* __restrict__ k,
                     const float* __restrict__ v, const float* __restrict__ ls,
                     const float* __restrict__ bias, float* __restrict__ out) {
  extern __shared__ char smem_raw[];
  _Float16* qs  = (_Float16*)smem_raw;            // 256 x 80 halves
  _Float16* ks  = qs  + NN * QK_STRIDE;           // 256 x 80 halves
  _Float16* vts = ks  + NN * QK_STRIDE;           // 64  x 272 halves (v transposed)
  _Float16* ptb = vts + DD * VT_STRIDE;           // 16 waves x 16 x 48 halves

  const int bh  = blockIdx.x;            // b*8 + h
  const int b   = bh >> 3;
  const int h   = bh & 7;
  const int tid = threadIdx.x;
  const size_t base = (size_t)bh * (NN * DD);

  // ---------------- Phase 1: stage operands into LDS as f16 ----------------
  // Threads 0..255: l2-normalize q rows. Threads 256..511: k rows.
  {
    const int row = tid & 255;
    const float* src = (tid < 256 ? q : k) + base + (size_t)row * DD;
    _Float16*   dst = (tid < 256 ? qs : ks) + (size_t)row * QK_STRIDE;
    v4f r[16];
    float s = 0.f;
#pragma unroll
    for (int i = 0; i < 16; ++i) {
      r[i] = ((const v4f*)src)[i];
      s += r[i].x * r[i].x + r[i].y * r[i].y + r[i].z * r[i].z + r[i].w * r[i].w;
    }
    const float rn = 1.0f / fmaxf(__builtin_sqrtf(s), 1e-12f);  // F.normalize eps
#pragma unroll
    for (int i = 0; i < 16; ++i) {
      v4h o;
      o.x = (_Float16)(r[i].x * rn); o.y = (_Float16)(r[i].y * rn);
      o.z = (_Float16)(r[i].z * rn); o.w = (_Float16)(r[i].w * rn);
      ((v4h*)dst)[i] = o;
    }
  }
  // All 512 threads: transpose+convert v (256x64 f32 -> vts[64][256] f16).
  {
    const int row = tid & 255;              // v row (will be a column of vts)
    const int c0  = (tid >> 8) * 32;        // 32 consecutive v columns
    const float* src = v + base + (size_t)row * DD + c0;
    v4f r[8];
#pragma unroll
    for (int i = 0; i < 8; ++i) r[i] = ((const v4f*)src)[i];
#pragma unroll
    for (int i = 0; i < 8; ++i) {
      vts[(size_t)(c0 + 4*i + 0) * VT_STRIDE + row] = (_Float16)r[i].x;
      vts[(size_t)(c0 + 4*i + 1) * VT_STRIDE + row] = (_Float16)r[i].y;
      vts[(size_t)(c0 + 4*i + 2) * VT_STRIDE + row] = (_Float16)r[i].z;
      vts[(size_t)(c0 + 4*i + 3) * VT_STRIDE + row] = (_Float16)r[i].w;
    }
  }
  __syncthreads();

  // ---------------- Phase 2: per-wave 16-row attention strip ----------------
  const int wave = tid >> 5;          // 0..15
  const int lane = tid & 31;
  const int lhi  = lane >> 4;         // lane-half (ISA WMMA layouts)
  const int llo  = lane & 15;
  const int m0   = wave * 16;         // first q row of this wave

  const float scale = __expf(fminf(ls[h], LOG100));

  // A fragments: q rows [m0, m0+16), K-chunks 0..31 and 32..63.
  // A 16x32 operand, k order used here (matches B): lane L: M=L%16; halves i=0..15 -> K = 16*lhi + i.
  Frag aq0, aq1;
  {
    const _Float16* qrow = qs + (size_t)(m0 + llo) * QK_STRIDE;
    const int kb = lhi * 16;
    aq0.h8[0] = *(const v8h*)(qrow + kb);
    aq0.h8[1] = *(const v8h*)(qrow + 8 + kb);
    aq1.h8[0] = *(const v8h*)(qrow + 32 + kb);
    aq1.h8[1] = *(const v8h*)(qrow + 40 + kb);
  }

  // S = qn @ kn^T : 16 column tiles of 16x16, each = 2 chained WMMAs (K=64).
  // B 32x16 operand, same k order as A: lane L: N=L%16; halves i=0..15 -> K = 16*lhi + i
  // (contiguous along a kn row since B = kn^T).
  v8f acc[16];
#pragma unroll
  for (int j = 0; j < 16; ++j) {
    const _Float16* krow = ks + (size_t)(j * 16 + llo) * QK_STRIDE;
    const v16h bk0 = *(const v16h*)(krow + lhi * 16);
    const v16h bk1 = *(const v16h*)(krow + 32 + lhi * 16);
    v8f c = {};
    c = __builtin_amdgcn_wmma_f32_16x16x32_f16(false, aq0.v, false, bk0,
                                               (short)0, c, false, false);
    c = __builtin_amdgcn_wmma_f32_16x16x32_f16(false, aq1.v, false, bk1,
                                               (short)0, c, false, false);
    acc[j] = c;
  }

  // scores = scores*scale + bias.  C layout: VGPR r -> (M = r + 8*lhi, N = llo).
  {
    const float* brow = bias + ((size_t)h * NN + (m0 + lhi * 8)) * NN + llo;
#pragma unroll
    for (int j = 0; j < 16; ++j)
#pragma unroll
      for (int r = 0; r < 8; ++r)
        acc[j][r] = acc[j][r] * scale + brow[(size_t)r * NN + j * 16];
  }

  // Row softmax. A row lives in one 16-lane half (xor offsets 1,2,4,8 stay
  // inside the half), one VGPR slot per row, across the 16 column tiles.
  {
    float mx[8], sm[8];
#pragma unroll
    for (int r = 0; r < 8; ++r) {
      mx[r] = acc[0][r];
#pragma unroll
      for (int j = 1; j < 16; ++j) mx[r] = fmaxf(mx[r], acc[j][r]);
      mx[r] = fmaxf(mx[r], __shfl_xor(mx[r], 8, 32));
      mx[r] = fmaxf(mx[r], __shfl_xor(mx[r], 4, 32));
      mx[r] = fmaxf(mx[r], __shfl_xor(mx[r], 2, 32));
      mx[r] = fmaxf(mx[r], __shfl_xor(mx[r], 1, 32));
      sm[r] = 0.f;
    }
#pragma unroll
    for (int j = 0; j < 16; ++j)
#pragma unroll
      for (int r = 0; r < 8; ++r) {
        const float p = __expf(acc[j][r] - mx[r]);
        acc[j][r] = p;
        sm[r] += p;
      }
#pragma unroll
    for (int r = 0; r < 8; ++r) {
      sm[r] += __shfl_xor(sm[r], 8, 32);
      sm[r] += __shfl_xor(sm[r], 4, 32);
      sm[r] += __shfl_xor(sm[r], 2, 32);
      sm[r] += __shfl_xor(sm[r], 1, 32);
      sm[r] = 1.0f / sm[r];
    }
#pragma unroll
    for (int j = 0; j < 16; ++j)
#pragma unroll
      for (int r = 0; r < 8; ++r) acc[j][r] *= sm[r];
  }

  // O = P @ V. Stage P through per-wave LDS scratch to convert register
  // C-layout -> WMMA A-layout (cross-lane move). Same-wave LDS ops are
  // in-order, so no barrier needed; only this wave touches its region.
  _Float16* pt = ptb + (size_t)wave * 16 * PT_STRIDE;
  v8f o0 = {}, o1 = {}, o2 = {}, o3 = {};
#pragma unroll
  for (int c = 0; c < 8; ++c) {                 // K-chunk = P cols [32c,32c+32)
#pragma unroll
    for (int jj = 0; jj < 2; ++jj) {
      const int j = 2 * c + jj;
      _Float16* prow = pt + (size_t)(lhi * 8) * PT_STRIDE + jj * 16 + llo;
#pragma unroll
      for (int r = 0; r < 8; ++r)
        prow[(size_t)r * PT_STRIDE] = (_Float16)acc[j][r];
    }
    Frag pa;                                    // A-layout read of P chunk
    {
      const _Float16* par = pt + (size_t)llo * PT_STRIDE;
      pa.h8[0] = *(const v8h*)(par + lhi * 16);
      pa.h8[1] = *(const v8h*)(par + 8 + lhi * 16);
    }
    // B fragments from v^T: contiguous 32B per lane.
    const _Float16* vb = vts + (size_t)llo * VT_STRIDE + c * 32 + lhi * 16;
    const v16h bv0 = *(const v16h*)(vb + (size_t)0  * 16 * VT_STRIDE);
    const v16h bv1 = *(const v16h*)(vb + (size_t)1  * 16 * VT_STRIDE);
    const v16h bv2 = *(const v16h*)(vb + (size_t)2  * 16 * VT_STRIDE);
    const v16h bv3 = *(const v16h*)(vb + (size_t)3  * 16 * VT_STRIDE);
    o0 = __builtin_amdgcn_wmma_f32_16x16x32_f16(false, pa.v, false, bv0,
                                                (short)0, o0, false, false);
    o1 = __builtin_amdgcn_wmma_f32_16x16x32_f16(false, pa.v, false, bv1,
                                                (short)0, o1, false, false);
    o2 = __builtin_amdgcn_wmma_f32_16x16x32_f16(false, pa.v, false, bv2,
                                                (short)0, o2, false, false);
    o3 = __builtin_amdgcn_wmma_f32_16x16x32_f16(false, pa.v, false, bv3,
                                                (short)0, o3, false, false);
  }

  // Store in reference's transposed layout: out[((b*N + n)*nH + h)*d + dd],
  // n = m0 + 8*lhi + r, dd = nt*16 + llo. 64B-coalesced per half-wave.
  float* orow = out + (((size_t)b * NN + (m0 + lhi * 8)) * NH + h) * DD + llo;
#pragma unroll
  for (int r = 0; r < 8; ++r) {
    orow[(size_t)r * NH * DD +  0] = o0[r];
    orow[(size_t)r * NH * DD + 16] = o1[r];
    orow[(size_t)r * NH * DD + 32] = o2[r];
    orow[(size_t)r * NH * DD + 48] = o3[r];
  }
}

extern "C" void kernel_launch(void* const* d_in, const int* in_sizes, int n_in,
                              void* d_out, int out_size, void* d_ws, size_t ws_size,
                              hipStream_t stream) {
  (void)in_sizes; (void)n_in; (void)out_size; (void)d_ws; (void)ws_size;
  const float* q    = (const float*)d_in[0];
  const float* k    = (const float*)d_in[1];
  const float* v    = (const float*)d_in[2];
  const float* ls   = (const float*)d_in[3];
  const float* bias = (const float*)d_in[4];
  float* out = (float*)d_out;
  attn_cos_kernel<<<2048, 512, SMEM_BYTES, stream>>>(q, k, v, ls, bias, out);
}
